// AngularAttention_58531814310184
// MI455X (gfx1250) — matrix operand order repaired and checked by reading, not compiled, not run on hardware
//
#include <hip/hip_runtime.h>
#include <hip/hip_bf16.h>
#include <math.h>

// ---------------------------------------------------------------------------
// Types for CDNA5 WMMA (wave32): v_wmma_f32_16x16x32_bf16
// ---------------------------------------------------------------------------
typedef __bf16 bf16_t;
typedef bf16_t bfvec16 __attribute__((ext_vector_type(16)));
typedef float  fvec8   __attribute__((ext_vector_type(8)));
typedef unsigned int u32x4 __attribute__((ext_vector_type(4)));
typedef int i32x8 __attribute__((ext_vector_type(8)));
typedef int i32x4 __attribute__((ext_vector_type(4)));

union FragAB {
    bfvec16  v;
    unsigned u[8];   // 8 x (2 packed bf16)
};

__device__ __forceinline__ unsigned short f2bf(float a) {
    unsigned ua = __builtin_bit_cast(unsigned, a);
    return (unsigned short)((ua + 0x7FFFu + ((ua >> 16) & 1u)) >> 16);
}
__device__ __forceinline__ unsigned pack_bf16x2(float a, float b) {
    unsigned ua = __builtin_bit_cast(unsigned, a);
    unsigned ub = __builtin_bit_cast(unsigned, b);
    ua = (ua + 0x7FFFu + ((ua >> 16) & 1u)) >> 16;
    ub = (ub + 0x7FFFu + ((ub >> 16) & 1u)) >> 16;
    return (ub << 16) | (ua & 0xFFFFu);
}

// A-fragment (16x32 bf16) from row-major [rows][lda].
// ISA 7.12.2: lane r=lane&15 -> row M=r; half h=lane>>4;
// VGPR i packs K = k0 + (i/4)*16 + h*8 + (i%4)*2.
__device__ __forceinline__ FragAB load_a(const unsigned short* __restrict__ A,
                                         int row0, int k0, int lda, int lane) {
    FragAB f;
    const int r = lane & 15, h = lane >> 4;
    const unsigned short* base = A + (size_t)(row0 + r) * lda + k0 + (h << 3);
#pragma unroll
    for (int i = 0; i < 8; ++i) {
        const int k = ((i >> 2) << 4) + ((i & 3) << 1);
        f.u[i] = *reinterpret_cast<const unsigned*>(base + k);
    }
    return f;
}

// B-fragment (32x16 bf16) from transposed source Bt row-major [N][ldb]
// (B[k][n] == Bt[n][k]).  ISA: lane n=lane&15 is column N; VGPR i packs
// K = k0 + (i/4)*16 + h*8 + (i%4)*2.
__device__ __forceinline__ FragAB load_bt(const unsigned short* __restrict__ Bt,
                                          int n0, int k0, int ldb, int lane) {
    FragAB f;
    const int n = lane & 15, h = lane >> 4;
    const unsigned short* base = Bt + (size_t)(n0 + n) * ldb + k0 + (h << 3);
#pragma unroll
    for (int i = 0; i < 8; ++i)
        f.u[i] = *reinterpret_cast<const unsigned*>(base + ((i >> 2) << 4) + ((i & 3) << 1));
    return f;
}

__device__ __forceinline__ fvec8 wmma_bf16(FragAB a, FragAB b, fvec8 c) {
    return __builtin_amdgcn_wmma_f32_16x16x32_bf16(
        false, a.v, false, b.v, (short)0, c, false, false);
}

// ---------------------------------------------------------------------------
// Tensor Data Mover: 2D tile Global -> LDS (ISA ch.8 D# descriptor).
// tensor dims == tile dims (tile always fully in-bounds -> no OOB path).
// Toolchain probe: this clang exposes the 6-arg builtin
//   (u32x4 g0, i32x8 g1, i32x4 g2, i32x4 g3, i32x8 extra, i32 cpol)
// ---------------------------------------------------------------------------
#if __has_builtin(__builtin_amdgcn_tensor_load_to_lds) && \
    __has_builtin(__builtin_amdgcn_s_wait_tensorcnt)
#define HAVE_TDM 1
#else
#define HAVE_TDM 0
#endif

__device__ __forceinline__ unsigned lds_off32(const void* p) {
    // generic pointer to LDS: addr[31:0] is the LDS byte offset (ISA 10.2)
    return (unsigned)(unsigned long long)p;
}

#if HAVE_TDM
__device__ __forceinline__ void tdm_load_2d(unsigned lds_addr, const void* gptr,
                                            unsigned tile_d0, unsigned tile_d1,
                                            unsigned stride0_elems) {
    unsigned long long ga = (unsigned long long)gptr;
    u32x4 g0;
    g0.x = 1u;                                   // count=1, user descriptor
    g0.y = lds_addr;                             // LDS byte address
    g0.z = (unsigned)ga;                         // global_addr[31:0]
    g0.w = (unsigned)((ga >> 32) & 0x01FFFFFFu)  // global_addr[56:32]
         | (2u << 30);                           // type = 2 ("image")
    i32x8 g1;
    g1[0] = 0x10000;                             // data_size = 1 (2 bytes/elem)
    g1[1] = (int)((tile_d0 & 0xFFFFu) << 16);    // tensor_dim0[15:0]
    g1[2] = (int)((tile_d0 >> 16) | ((tile_d1 & 0xFFFFu) << 16)); // d0 hi, d1 lo
    g1[3] = (int)((tile_d1 >> 16) | (tile_d0 << 16));             // d1 hi, tile_dim0
    g1[4] = (int)(tile_d1 & 0xFFFFu);            // tile_dim1 (tile_dim2 = 0)
    g1[5] = (int)stride0_elems;                  // tensor_dim0_stride[31:0]
    g1[6] = 0;
    g1[7] = 0;
    i32x4 z4 = {0, 0, 0, 0};                     // 2D: groups 2/3 unused
    i32x8 z8 = {0, 0, 0, 0, 0, 0, 0, 0};
    __builtin_amdgcn_tensor_load_to_lds(g0, g1, z4, z4, z8, 0);
}
#endif

// Fallback cooperative copy (u32 granularity), used when TDM builtin absent.
__device__ __forceinline__ void coop_copy_u32(unsigned* __restrict__ dst,
                                              const unsigned* __restrict__ src,
                                              int n_u32, int tid, int nthr) {
    for (int i = tid; i < n_u32; i += nthr) dst[i] = src[i];
}

// ---------------------------------------------------------------------------
// Problem constants (B=2, T=2048, d=1024, H=16, dk=64)
// ---------------------------------------------------------------------------
#define DM   1024
#define TT   2048
#define NB   2
#define NH   16
#define DK   64
#define MROW (NB * TT)   // 4096

// ---------------------------------------------------------------------------
// Kernel 0: f32 -> bf16 conversion
// ---------------------------------------------------------------------------
__global__ void __launch_bounds__(256)
cvt_kernel(const float* __restrict__ src, unsigned short* __restrict__ dst, int nvec) {
    int idx = blockIdx.x * blockDim.x + threadIdx.x;
    if (idx < nvec) {
        float2 v = reinterpret_cast<const float2*>(src)[idx];
        reinterpret_cast<unsigned*>(dst)[idx] = pack_bf16x2(v.x, v.y);
    }
}

// ---------------------------------------------------------------------------
// Kernel 1: fused QKV projection + L2 normalize.
// Block = 8 waves sharing one 16-row x-tile (staged 32KB in LDS via TDM).
// Wave -> 16x64 tile (one head slice of one of Q/K/V).
// ---------------------------------------------------------------------------
__global__ void __launch_bounds__(256)
qkv_kernel(const unsigned short* __restrict__ xbf,
           const unsigned short* __restrict__ Wqbf,
           const unsigned short* __restrict__ Wkbf,
           const unsigned short* __restrict__ Wvbf,
           unsigned short* __restrict__ Qn,
           unsigned short* __restrict__ Kn,
           unsigned short* __restrict__ Vt) {
    __shared__ unsigned short xtile[16 * DM];    // 32 KB

    const int lane  = threadIdx.x & 31;
    const int wv    = threadIdx.x >> 5;
    const int wg    = blockIdx.x * 8 + wv;
    const int which = wg % 3;                    // 0=Q 1=K 2=V
    const int head  = (wg / 3) & (NH - 1);
    const int mtile = wg / (3 * NH);             // 0..255 (uniform per block)
    const int row0  = mtile * 16;

    // stage A-tile: x[row0:row0+16, 0:1024] -> LDS
#if HAVE_TDM
    if (wv == 0) {
        tdm_load_2d(lds_off32(xtile), xbf + (size_t)row0 * DM, DM, 16, DM);
        __builtin_amdgcn_s_wait_tensorcnt(0);
    }
#else
    coop_copy_u32((unsigned*)xtile, (const unsigned*)(xbf + (size_t)row0 * DM),
                  16 * DM / 2, threadIdx.x, 256);
#endif
    __syncthreads();

    const unsigned short* W = (which == 0) ? Wqbf : (which == 1) ? Wkbf : Wvbf;

    fvec8 acc[4];
#pragma unroll
    for (int s = 0; s < 4; ++s) acc[s] = (fvec8)(0.0f);

    for (int k0 = 0; k0 < DM; k0 += 32) {
        __builtin_prefetch(W + (size_t)(head * DK + (lane & 15)) * DM + k0 + 64, 0, 0);
        FragAB a = load_a(xtile, 0, k0, DM, lane);
        FragAB b[4];
#pragma unroll
        for (int s = 0; s < 4; ++s)
            b[s] = load_bt(W, head * DK + s * 16, k0, DM, lane);
#pragma unroll
        for (int s = 0; s < 4; ++s)
            acc[s] = wmma_bf16(a, b[s], acc[s]);
    }

    const int half = lane >> 4, n = lane & 15;

    if (which < 2) {  // L2 normalize rows over dk=64
#pragma unroll
        for (int i = 0; i < 8; ++i) {
            float p = 0.f;
#pragma unroll
            for (int s = 0; s < 4; ++s) p += acc[s][i] * acc[s][i];
#pragma unroll
            for (int off = 1; off < 16; off <<= 1) p += __shfl_xor(p, off, 32);
            float scale = 1.0f / fmaxf(__builtin_sqrtf(p), 1e-12f);
#pragma unroll
            for (int s = 0; s < 4; ++s) acc[s][i] *= scale;
        }
    }

    const int b     = row0 >> 11;
    const int tbase = row0 & (TT - 1);

    if (which < 2) {
        unsigned short* dst =
            ((which == 0) ? Qn : Kn) + (size_t)(b * NH + head) * TT * DK;
#pragma unroll
        for (int s = 0; s < 4; ++s)
#pragma unroll
            for (int i = 0; i < 8; ++i)
                dst[(size_t)(tbase + i + 8 * half) * DK + s * 16 + n] = f2bf(acc[s][i]);
    } else {
        unsigned short* dst = Vt + (size_t)(b * NH + head) * DK * TT;
#pragma unroll
        for (int s = 0; s < 4; ++s)
#pragma unroll
            for (int i = 0; i < 8; ++i)
                dst[(size_t)(s * 16 + n) * TT + tbase + i + 8 * half] = f2bf(acc[s][i]);
    }
}

// ---------------------------------------------------------------------------
// Kernel 2: attention.  Block = 8 waves on one (b,h), 8 q-tiles.
// K/V 32-key chunks double-buffered in LDS via TDM (wave 0 issues,
// s_wait_tensorcnt 2 overlaps DMA of chunk c+1 with compute of chunk c).
// ---------------------------------------------------------------------------
#define CHUNK 32
#define NCHUNK (TT / CHUNK)   // 64

__global__ void __launch_bounds__(256)
attn_kernel(const unsigned short* __restrict__ Qn,
            const unsigned short* __restrict__ Kn,
            const unsigned short* __restrict__ Vt,
            unsigned short* __restrict__ attn) {
    __shared__ unsigned short ktile[2][CHUNK * DK];   // 2 x 4 KB  [t][c]
    __shared__ unsigned short vtile[2][DK * CHUNK];   // 2 x 4 KB  [c][t]
    __shared__ unsigned short wtile[8][16 * CHUNK];   // 8 x 1 KB per-wave

    const int lane = threadIdx.x & 31;
    const int wv   = threadIdx.x >> 5;
    const int wg   = blockIdx.x * 8 + wv;
    const int qt   = wg & 127;            // q-tile (uniform bh per block)
    const int bh   = wg >> 7;             // 0..31

    const unsigned short* Qh = Qn + (size_t)bh * TT * DK;
    const unsigned short* Kh = Kn + (size_t)bh * TT * DK;
    const unsigned short* Vh = Vt + (size_t)bh * DK * TT;
    unsigned short* lw = wtile[wv];

    const int half = lane >> 4, n = lane & 15;

    fvec8 o[4];
#pragma unroll
    for (int s = 0; s < 4; ++s) o[s] = (fvec8)(0.0f);
    float den[8];
#pragma unroll
    for (int i = 0; i < 8; ++i) den[i] = 0.f;

    FragAB aq0 = load_a(Qh, qt * 16, 0, DK, lane);
    FragAB aq1 = load_a(Qh, qt * 16, 32, DK, lane);

    const float inv_pi = 0.3183098861837907f;

#if HAVE_TDM
    if (wv == 0) {  // prologue: chunk 0 -> buffer 0
        tdm_load_2d(lds_off32(ktile[0]), Kh, DK, CHUNK, DK);
        tdm_load_2d(lds_off32(vtile[0]), Vh, CHUNK, DK, TT);
    }
#endif

    for (int c = 0; c < NCHUNK; ++c) {
        const int kt = c * CHUNK;
#if HAVE_TDM
        const int p = c & 1;
        if (wv == 0) {
            if (c + 1 < NCHUNK) {
                tdm_load_2d(lds_off32(ktile[p ^ 1]),
                            Kh + (size_t)(kt + CHUNK) * DK, DK, CHUNK, DK);
                tdm_load_2d(lds_off32(vtile[p ^ 1]),
                            Vh + (size_t)(kt + CHUNK), CHUNK, DK, TT);
                __builtin_amdgcn_s_wait_tensorcnt(2);  // chunk c complete
            } else {
                __builtin_amdgcn_s_wait_tensorcnt(0);
            }
        }
        __syncthreads();   // publish buffer p to all waves
#else
        const int p = 0;
        __syncthreads();
        coop_copy_u32((unsigned*)ktile[0],
                      (const unsigned*)(Kh + (size_t)kt * DK),
                      CHUNK * DK / 2, threadIdx.x, 256);
        // strided V rows: 64 rows of 32 elems, row stride TT
        for (int idx = threadIdx.x; idx < DK * (CHUNK / 2); idx += 256) {
            int r = idx / (CHUNK / 2), t2 = idx % (CHUNK / 2);
            ((unsigned*)vtile[0])[r * (CHUNK / 2) + t2] =
                *(const unsigned*)(Vh + (size_t)r * TT + kt + 2 * t2);
        }
        __syncthreads();
#endif

        // S = Q.K^T for two 16-key sub-tiles (dk = 64 -> 2 chained WMMAs each)
        fvec8 s0 = (fvec8)(0.0f), s1 = (fvec8)(0.0f);
        {
            FragAB b00 = load_bt(ktile[p], 0,  0,  DK, lane);
            FragAB b01 = load_bt(ktile[p], 0,  32, DK, lane);
            FragAB b10 = load_bt(ktile[p], 16, 0,  DK, lane);
            FragAB b11 = load_bt(ktile[p], 16, 32, DK, lane);
            s0 = wmma_bf16(aq0, b00, s0);
            s0 = wmma_bf16(aq1, b01, s0);
            s1 = wmma_bf16(aq0, b10, s1);
            s1 = wmma_bf16(aq1, b11, s1);
        }

        // clip -> 1-acos/pi -> max(.,eps)^8 ; accumulate denominator; bf16 tile
#pragma unroll
        for (int i = 0; i < 8; ++i) {
            const int m = i + 8 * half;
            float sim0 = fminf(fmaxf(s0[i], -0.999f), 0.999f);
            float sim1 = fminf(fmaxf(s1[i], -0.999f), 0.999f);
            float w0 = fmaxf(1.0f - acosf(sim0) * inv_pi, 1e-6f);
            float w1 = fmaxf(1.0f - acosf(sim1) * inv_pi, 1e-6f);
            w0 = w0 * w0; w0 = w0 * w0; w0 = w0 * w0;   // ^8
            w1 = w1 * w1; w1 = w1 * w1; w1 = w1 * w1;
            den[i] += w0 + w1;
            lw[m * CHUNK + n]      = f2bf(w0);
            lw[m * CHUNK + 16 + n] = f2bf(w1);
        }
        // intra-wave LDS round-trip: DS ops are in-order within a wave,
        // lanes are lockstep -> no barrier needed for the private wtile.

        // O += W.V  (V^T tile gives contiguous k-pairs)
        FragAB aw = load_a(lw, 0, 0, CHUNK, lane);
#pragma unroll
        for (int s = 0; s < 4; ++s) {
            FragAB bv = load_bt(vtile[p], s * 16, 0, CHUNK, lane);
            o[s] = wmma_bf16(aw, bv, o[s]);
        }
        __syncthreads();   // all waves done with buffer p before TDM reuses it
    }

    // out = num / (den + eps)
#pragma unroll
    for (int i = 0; i < 8; ++i) {
        float psum = den[i];
#pragma unroll
        for (int off = 1; off < 16; off <<= 1) psum += __shfl_xor(psum, off, 32);
        float inv = 1.0f / (psum + 1e-6f);
#pragma unroll
        for (int s = 0; s < 4; ++s) o[s][i] *= inv;
    }

    const int browq = (bh >> 4) * TT + qt * 16;
    const int col0  = (bh & 15) * DK;
#pragma unroll
    for (int s = 0; s < 4; ++s)
#pragma unroll
        for (int i = 0; i < 8; ++i)
            attn[(size_t)(browq + i + 8 * half) * DM + col0 + s * 16 + n] =
                f2bf(o[s][i]);
}

// ---------------------------------------------------------------------------
// Kernel 3: output projection  y = attn @ Wo^T + bo  (f32 out).
// Block = 8 waves sharing one 16-row A-tile staged via TDM.
// ---------------------------------------------------------------------------
__global__ void __launch_bounds__(256)
oproj_kernel(const unsigned short* __restrict__ attn,
             const unsigned short* __restrict__ Wobf,
             const float* __restrict__ bo,
             float* __restrict__ out) {
    __shared__ unsigned short atile[16 * DM];    // 32 KB

    const int lane  = threadIdx.x & 31;
    const int wv    = threadIdx.x >> 5;
    const int wg    = blockIdx.x * 8 + wv;
    const int ntile = wg & 15;
    const int mtile = wg >> 4;                   // uniform per block

#if HAVE_TDM
    if (wv == 0) {
        tdm_load_2d(lds_off32(atile), attn + (size_t)mtile * 16 * DM, DM, 16, DM);
        __builtin_amdgcn_s_wait_tensorcnt(0);
    }
#else
    coop_copy_u32((unsigned*)atile,
                  (const unsigned*)(attn + (size_t)mtile * 16 * DM),
                  16 * DM / 2, threadIdx.x, 256);
#endif
    __syncthreads();

    fvec8 acc[4];
#pragma unroll
    for (int s = 0; s < 4; ++s) acc[s] = (fvec8)(0.0f);

    for (int k0 = 0; k0 < DM; k0 += 32) {
        __builtin_prefetch(Wobf + (size_t)(ntile * 64 + (lane & 15)) * DM + k0 + 64, 0, 0);
        FragAB a = load_a(atile, 0, k0, DM, lane);
        FragAB b[4];
#pragma unroll
        for (int s = 0; s < 4; ++s)
            b[s] = load_bt(Wobf, ntile * 64 + s * 16, k0, DM, lane);
#pragma unroll
        for (int s = 0; s < 4; ++s)
            acc[s] = wmma_bf16(a, b[s], acc[s]);
    }

    const int half = lane >> 4, n = lane & 15;
#pragma unroll
    for (int s = 0; s < 4; ++s)
#pragma unroll
        for (int i = 0; i < 8; ++i) {
            int row = mtile * 16 + i + 8 * half;
            int col = ntile * 64 + s * 16 + n;
            out[(size_t)row * DM + col] = acc[s][i] + bo[col];
        }
}

// ---------------------------------------------------------------------------
// Host-side launcher
// ---------------------------------------------------------------------------
extern "C" void kernel_launch(void* const* d_in, const int* in_sizes, int n_in,
                              void* d_out, int out_size, void* d_ws, size_t ws_size,
                              hipStream_t stream) {
    (void)in_sizes; (void)n_in; (void)out_size; (void)ws_size;

    const float* x  = (const float*)d_in[0];
    const float* Wq = (const float*)d_in[1];
    const float* Wk = (const float*)d_in[2];
    const float* Wv = (const float*)d_in[3];
    const float* Wo = (const float*)d_in[4];
    const float* bo = (const float*)d_in[5];
    float* out = (float*)d_out;

    unsigned short* ws = (unsigned short*)d_ws;
    unsigned short* xbf  = ws;                         // 4096*1024
    unsigned short* Wqbf = xbf  + (size_t)MROW * DM;
    unsigned short* Wkbf = Wqbf + (size_t)DM * DM;
    unsigned short* Wvbf = Wkbf + (size_t)DM * DM;
    unsigned short* Wobf = Wvbf + (size_t)DM * DM;
    unsigned short* Qn   = Wobf + (size_t)DM * DM;     // [B,H,T,64]
    unsigned short* Kn   = Qn   + (size_t)MROW * DM;
    unsigned short* Vt   = Kn   + (size_t)MROW * DM;   // [B,H,64,T]
    unsigned short* attn = Vt   + (size_t)MROW * DM;   // [4096,1024]

    {
        int nv_x = (MROW * DM) / 2;
        int nv_w = (DM * DM) / 2;
        cvt_kernel<<<(nv_x + 255) / 256, 256, 0, stream>>>(x,  xbf,  nv_x);
        cvt_kernel<<<(nv_w + 255) / 256, 256, 0, stream>>>(Wq, Wqbf, nv_w);
        cvt_kernel<<<(nv_w + 255) / 256, 256, 0, stream>>>(Wk, Wkbf, nv_w);
        cvt_kernel<<<(nv_w + 255) / 256, 256, 0, stream>>>(Wv, Wvbf, nv_w);
        cvt_kernel<<<(nv_w + 255) / 256, 256, 0, stream>>>(Wo, Wobf, nv_w);
    }

    qkv_kernel<<<12288 / 8, 256, 0, stream>>>(xbf, Wqbf, Wkbf, Wvbf, Qn, Kn, Vt);
    attn_kernel<<<4096 / 8, 256, 0, stream>>>(Qn, Kn, Vt, attn);
    oproj_kernel<<<4096 / 8, 256, 0, stream>>>(attn, Wobf, bo, out);
}
